// SentAttNet_82549271429306
// MI455X (gfx1250) — matrix operand order repaired and checked by reading, not compiled, not run on hardware
//
#include <hip/hip_runtime.h>

// ---------------------------------------------------------------------------
// SentAttNet on MI455X (gfx1250): bf16 WMMA GRU + attention + classifier.
// Round 2: software-pipelined x-tile loads in the recurrent kernel (register
// double-buffer + global_prefetch_b8 for t+2) to take HBM latency off the
// 128-step serial dependency chain.
// ---------------------------------------------------------------------------

typedef __attribute__((ext_vector_type(16))) __bf16 v16bf;
typedef __attribute__((ext_vector_type(8)))  float  v8f;

#define SEQ 128
#define BATCH 4096
#define HID 50
#define INP 100          // 2*word_hidden
#define DD 100           // 2*HID (bidir output dim)
#define NC 14
#define N3 150           // 3*HID (gates r|z|n)
#define NP 160           // N padded to 10 WMMA tiles
#define KIP 128          // I padded (K for input GEMM)
#define KHP 64           // H padded (K for hidden GEMM)
#define MT 32            // batch rows per block
#define DP 112           // DD padded to 7 WMMA tiles

__device__ __forceinline__ float sigmoidf(float v) {
    return 1.0f / (1.0f + __expf(-v));
}

// ---------------------------------------------------------------------------
// Kernel 1: fused bidirectional GRU. grid = (BATCH/MT, 2 dirs), block = 256.
// Each block owns 32 batch rows of one direction for all 128 steps.
// ---------------------------------------------------------------------------
__global__ __launch_bounds__(256) void gru_kernel(
    const float* __restrict__ x,      const float* __restrict__ h0,
    const float* __restrict__ wih_f,  const float* __restrict__ whh_f,
    const float* __restrict__ bih_f,  const float* __restrict__ bhh_f,
    const float* __restrict__ wih_b,  const float* __restrict__ whh_b,
    const float* __restrict__ bih_b,  const float* __restrict__ bhh_b,
    float* __restrict__ fout,         // [SEQ][BATCH][2*HID]
    float* __restrict__ outH)         // [2][BATCH][HID]
{
    __shared__ __bf16 wihL[NP * KIP];   // Wih^T zero-padded: [n][k] = wih[n][k]
    __shared__ __bf16 whhL[NP * KHP];   // Whh^T zero-padded
    __shared__ __bf16 xA[MT * KIP];     // x_t tile, bf16, zero-padded K
    __shared__ __bf16 hA[MT * KHP];     // h tile, bf16, zero-padded K
    __shared__ float  giL[MT * NP];     // staged input pre-activations
    __shared__ float  ghL[MT * NP];     // staged hidden pre-activations
    __shared__ float  hF[MT * HID];     // running hidden state (fp32)
    __shared__ float  biasL[2 * N3];    // bih ++ bhh

    const int tid = threadIdx.x;
    const int dir = blockIdx.y;
    const int rowBase = blockIdx.x * MT;

    const float* wih = dir ? wih_b : wih_f;
    const float* whh = dir ? whh_b : whh_f;
    const float* bih = dir ? bih_b : bih_f;
    const float* bhh = dir ? bhh_b : bhh_f;

    // One-time: weights -> LDS bf16 (zero padded), biases, h0.
    for (int idx = tid; idx < NP * KIP; idx += 256) {
        int n = idx / KIP, k = idx % KIP;
        wihL[idx] = (n < N3 && k < INP) ? (__bf16)wih[n * INP + k] : (__bf16)0.0f;
    }
    for (int idx = tid; idx < NP * KHP; idx += 256) {
        int n = idx / KHP, k = idx % KHP;
        whhL[idx] = (n < N3 && k < HID) ? (__bf16)whh[n * HID + k] : (__bf16)0.0f;
    }
    for (int idx = tid; idx < N3; idx += 256) {
        biasL[idx]      = bih[idx];
        biasL[N3 + idx] = bhh[idx];
    }
    for (int idx = tid; idx < MT * HID; idx += 256) {
        int r = idx / HID, j = idx % HID;
        hF[idx] = h0[(size_t)dir * BATCH * HID + (size_t)(rowBase + r) * HID + j];
    }

    const int wave = tid >> 5;
    const int lane = tid & 31;
    const int half = lane >> 4;
    const int l16  = lane & 15;

    // Register pipeline for x tiles: 16 slots cover MT*KIP / 256 threads.
    float xreg[16];
    {   // prologue: load x for step s = 0
        const int t0 = dir ? (SEQ - 1) : 0;
        const float* xt = x + ((size_t)t0 * BATCH + rowBase) * INP;
        #pragma unroll
        for (int i = 0; i < 16; ++i) {
            const int idx = tid + i * 256;
            const int r = idx / KIP, k = idx % KIP;
            xreg[i] = (k < INP) ? xt[(size_t)r * INP + k] : 0.0f;
        }
    }
    __syncthreads();

    for (int s = 0; s < SEQ; ++s) {
        const int t = dir ? (SEQ - 1 - s) : s;

        // Stage x_t (from register pipeline) and h as bf16 A-operands.
        #pragma unroll
        for (int i = 0; i < 16; ++i) {
            xA[tid + i * 256] = (__bf16)xreg[i];
        }
        for (int idx = tid; idx < MT * KHP; idx += 256) {
            int r = idx / KHP, k = idx % KHP;
            hA[idx] = (k < HID) ? (__bf16)hF[r * HID + k] : (__bf16)0.0f;
        }
        __syncthreads();

        // Issue next step's x loads now; latency overlaps WMMA + gate phases.
        if (s + 1 < SEQ) {
            const int tn = dir ? (SEQ - 2 - s) : (s + 1);
            const float* xt = x + ((size_t)tn * BATCH + rowBase) * INP;
            #pragma unroll
            for (int i = 0; i < 16; ++i) {
                const int idx = tid + i * 256;
                const int r = idx / KIP, k = idx % KIP;
                xreg[i] = (k < INP) ? xt[(size_t)r * INP + k] : 0.0f;
            }
        }
        // Prefetch t+2 into cache (global_prefetch_b8).
        if (s + 2 < SEQ) {
            const int tp = dir ? (SEQ - 3 - s) : (s + 2);
            const float* xp = x + ((size_t)tp * BATCH + rowBase) * INP + (size_t)tid * 16;
            __builtin_prefetch(xp, 0, 1);
        }

        // 2 M-tiles x 10 N-tiles = 20 output tiles across 8 waves.
        for (int tile = wave; tile < 20; tile += 8) {
            const int mt = tile / 10, nt = tile % 10;
            const int arow = mt * 16 + l16;
            const int n    = nt * 16 + l16;

            v8f accI, accH;
            #pragma unroll
            for (int v = 0; v < 8; ++v) { accI[v] = 0.0f; accH[v] = 0.0f; }

            #pragma unroll
            for (int ks = 0; ks < 4; ++ks) {          // K = 128 for gi
                v16bf a, bw;
                #pragma unroll
                for (int e = 0; e < 16; ++e)
                    a[e] = xA[arow * KIP + ks * 32 + ((e < 8) ? e : e + 8) + 8 * half];
                #pragma unroll
                for (int e = 0; e < 16; ++e)
                    bw[e] = wihL[n * KIP + ks * 32 + ((e < 8) ? e : e + 8) + 8 * half];
                accI = __builtin_amdgcn_wmma_f32_16x16x32_bf16(
                    false, a, false, bw, (short)0, accI, false, false);
            }
            #pragma unroll
            for (int ks = 0; ks < 2; ++ks) {          // K = 64 for gh
                v16bf a, bw;
                #pragma unroll
                for (int e = 0; e < 16; ++e)
                    a[e] = hA[arow * KHP + ks * 32 + ((e < 8) ? e : e + 8) + 8 * half];
                #pragma unroll
                for (int e = 0; e < 16; ++e)
                    bw[e] = whhL[n * KHP + ks * 32 + ((e < 8) ? e : e + 8) + 8 * half];
                accH = __builtin_amdgcn_wmma_f32_16x16x32_bf16(
                    false, a, false, bw, (short)0, accH, false, false);
            }
            #pragma unroll
            for (int v = 0; v < 8; ++v) {
                const int r = mt * 16 + v + 8 * half;
                giL[r * NP + n] = accI[v];
                ghL[r * NP + n] = accH[v];
            }
        }
        __syncthreads();

        // Gate math (PyTorch GRU): r,z sigmoid; n = tanh(i_n + r*h_n).
        float* foutT = fout + ((size_t)t * BATCH + rowBase) * DD + dir * HID;
        const bool finalStep = (s == SEQ - 1);
        for (int idx = tid; idx < MT * HID; idx += 256) {
            const int r = idx / HID, j = idx % HID;
            const float gr = giL[r * NP + j]           + biasL[j]
                           + ghL[r * NP + j]           + biasL[N3 + j];
            const float gz = giL[r * NP + HID + j]     + biasL[HID + j]
                           + ghL[r * NP + HID + j]     + biasL[N3 + HID + j];
            const float rr = sigmoidf(gr);
            const float zz = sigmoidf(gz);
            const float gn = giL[r * NP + 2 * HID + j] + biasL[2 * HID + j]
                           + rr * (ghL[r * NP + 2 * HID + j] + biasL[N3 + 2 * HID + j]);
            const float nn = tanhf(gn);
            const float hnew = (1.0f - zz) * nn + zz * hF[r * HID + j];
            hF[r * HID + j] = hnew;
            foutT[(size_t)r * DD + j] = hnew;
            if (finalStep)
                outH[(size_t)dir * BATCH * HID + (size_t)(rowBase + r) * HID + j] = hnew;
        }
        __syncthreads();
    }
}

// ---------------------------------------------------------------------------
// Kernel 2: attention scores. u = tanh(f_out @ W + b); score = tanh(u . cw).
// grid = S*B/32 blocks of 256 threads, WMMA over [32 x 100] x [100 x 100].
// ---------------------------------------------------------------------------
__global__ __launch_bounds__(256) void attn_score_kernel(
    const float* __restrict__ fout, const float* __restrict__ sw,
    const float* __restrict__ sb,   const float* __restrict__ cw,
    float* __restrict__ scores)
{
    __shared__ __bf16 swL[KIP * DP];   // [k][n] zero-padded (28.7 KB)
    __shared__ __bf16 fA[MT * KIP];    // row tile of f_out
    __shared__ float  uL[MT * DP];     // tanh-projected rows
    __shared__ float  sbL[DD];
    __shared__ float  cwL[DD];

    const int tid = threadIdx.x;
    const size_t rowBase = (size_t)blockIdx.x * MT;

    for (int idx = tid; idx < KIP * DP; idx += 256) {
        int k = idx / DP, n = idx % DP;
        swL[idx] = (k < DD && n < DD) ? (__bf16)sw[k * DD + n] : (__bf16)0.0f;
    }
    for (int idx = tid; idx < DD; idx += 256) {
        sbL[idx] = sb[idx];
        cwL[idx] = cw[idx];
    }
    const float* frow = fout + rowBase * DD;
    for (int idx = tid; idx < MT * KIP; idx += 256) {
        int r = idx / KIP, k = idx % KIP;
        fA[idx] = (k < DD) ? (__bf16)frow[(size_t)r * DD + k] : (__bf16)0.0f;
    }
    __syncthreads();

    const int wave = tid >> 5;
    const int lane = tid & 31;
    const int half = lane >> 4;
    const int l16  = lane & 15;

    for (int tile = wave; tile < 14; tile += 8) {      // 2 M x 7 N tiles
        const int mt = tile / 7, nt = tile % 7;
        const int arow = mt * 16 + l16;
        const int n    = nt * 16 + l16;
        v8f acc;
        #pragma unroll
        for (int v = 0; v < 8; ++v) acc[v] = 0.0f;
        #pragma unroll
        for (int ks = 0; ks < 4; ++ks) {
            v16bf a, bw;
            #pragma unroll
            for (int e = 0; e < 16; ++e)
                a[e] = fA[arow * KIP + ks * 32 + ((e < 8) ? e : e + 8) + 8 * half];
            #pragma unroll
            for (int e = 0; e < 16; ++e)
                bw[e] = swL[(ks * 32 + ((e < 8) ? e : e + 8) + 8 * half) * DP + n];
            acc = __builtin_amdgcn_wmma_f32_16x16x32_bf16(
                false, a, false, bw, (short)0, acc, false, false);
        }
        #pragma unroll
        for (int v = 0; v < 8; ++v) {
            const int r = mt * 16 + v + 8 * half;
            uL[r * DP + n] = (n < DD) ? tanhf(acc[v] + sbL[n]) : 0.0f;
        }
    }
    __syncthreads();

    if (tid < MT) {
        float sum = 0.0f;
        for (int d = 0; d < DD; ++d) sum += uL[tid * DP + d] * cwL[d];
        scores[rowBase + tid] = tanhf(sum);
    }
}

// ---------------------------------------------------------------------------
// Kernel 3: softmax over S, attention pooling, FC. One block per batch col.
// ---------------------------------------------------------------------------
__global__ __launch_bounds__(128) void pool_fc_kernel(
    const float* __restrict__ fout, const float* __restrict__ scores,
    const float* __restrict__ fcw,  const float* __restrict__ fcb,
    float* __restrict__ out)
{
    __shared__ float red[128];
    __shared__ float attnL[SEQ];
    __shared__ float pooledL[DD];

    const int b = blockIdx.x;
    const int tid = threadIdx.x;     // tid == s

    const float sc = scores[(size_t)tid * BATCH + b];
    red[tid] = sc;
    __syncthreads();
    for (int off = 64; off > 0; off >>= 1) {
        if (tid < off) red[tid] = fmaxf(red[tid], red[tid + off]);
        __syncthreads();
    }
    const float mx = red[0];
    __syncthreads();
    const float e = __expf(sc - mx);
    red[tid] = e;
    __syncthreads();
    for (int off = 64; off > 0; off >>= 1) {
        if (tid < off) red[tid] += red[tid + off];
        __syncthreads();
    }
    attnL[tid] = e / red[0];
    __syncthreads();

    if (tid < DD) {
        float acc = 0.0f;
        for (int s2 = 0; s2 < SEQ; ++s2)
            acc += attnL[s2] * fout[((size_t)s2 * BATCH + b) * DD + tid];
        pooledL[tid] = acc;
    }
    __syncthreads();

    if (tid < NC) {
        float acc = fcb[tid];
        for (int d = 0; d < DD; ++d) acc += pooledL[d] * fcw[tid * DD + d];
        out[(size_t)b * NC + tid] = acc;
    }
}

// ---------------------------------------------------------------------------
extern "C" void kernel_launch(void* const* d_in, const int* in_sizes, int n_in,
                              void* d_out, int out_size, void* d_ws, size_t ws_size,
                              hipStream_t stream) {
    (void)in_sizes; (void)n_in; (void)out_size; (void)ws_size;

    const float* x     = (const float*)d_in[0];
    const float* h0    = (const float*)d_in[1];
    const float* wih_f = (const float*)d_in[2];
    const float* whh_f = (const float*)d_in[3];
    const float* bih_f = (const float*)d_in[4];
    const float* bhh_f = (const float*)d_in[5];
    const float* wih_b = (const float*)d_in[6];
    const float* whh_b = (const float*)d_in[7];
    const float* bih_b = (const float*)d_in[8];
    const float* bhh_b = (const float*)d_in[9];
    const float* sw    = (const float*)d_in[10];
    const float* sb    = (const float*)d_in[11];
    const float* cw    = (const float*)d_in[12];
    const float* fcw   = (const float*)d_in[13];
    const float* fcb   = (const float*)d_in[14];

    float* out    = (float*)d_out;                       // [B,NC] then [2,B,H]
    float* outH   = out + (size_t)BATCH * NC;
    float* foutWs = (float*)d_ws;                        // [S,B,2H]
    float* scWs   = foutWs + (size_t)SEQ * BATCH * DD;   // [S,B]

    dim3 g1(BATCH / MT, 2);
    gru_kernel<<<g1, 256, 0, stream>>>(x, h0, wih_f, whh_f, bih_f, bhh_f,
                                       wih_b, whh_b, bih_b, bhh_b, foutWs, outH);
    attn_score_kernel<<<(SEQ * BATCH) / MT, 256, 0, stream>>>(foutWs, sw, sb, cw, scWs);
    pool_fc_kernel<<<BATCH, 128, 0, stream>>>(foutWs, scWs, fcw, fcb, out);
}
